// AdvancedGNN_66022237274495
// MI455X (gfx1250) — compile-verified, matrix operand order checked by reading, not run on hardware
//
#include <hip/hip_runtime.h>
#include <hip/hip_bf16.h>

typedef _Float16 v16h __attribute__((ext_vector_type(16)));
typedef _Float16 v8h  __attribute__((ext_vector_type(8)));
typedef float    v8f  __attribute__((ext_vector_type(8)));

#define NN     50000
#define EEDGE  400000
#define ETOT   450000   // EEDGE + NN self loops
#define FD     256
#define HH     4
#define CC     64
#define GG     64
#define NCLS   10
#define NEGSL  0.2f
#define BNEPS  1e-5f

// ---------- helpers ----------
__device__ __forceinline__ void atomicMaxF(float* addr, float v) {
    if (v >= 0.0f) atomicMax((int*)addr, __float_as_int(v));
    else           atomicMin((unsigned int*)addr, (unsigned int)__float_as_int(v));
}

__device__ __forceinline__ v16h load_frag16(const _Float16* p) {
    union { v16h v; v8h h[2]; } u;
    u.h[0] = *(const v8h*)(p);        // K = kb+0..7   (this half-wave's low chunk)
    u.h[1] = *(const v8h*)(p + 16);   // K = kb+16..23 (high chunk)
    return u.v;
}

// ---------- conversion kernels ----------
__global__ void k_fill(float* p, long n, float v) {
    long i = (long)blockIdx.x * blockDim.x + threadIdx.x;
    if (i < n) p[i] = v;
}

__global__ void k_f32_to_f16(const float* x, _Float16* y, long n) {
    long i = (long)blockIdx.x * blockDim.x + threadIdx.x;
    if (i < n) y[i] = (_Float16)x[i];
}

// W[256x256] row-major (K x N) -> Wt[256x256] f16 (N x K) so B-fragment loads are contiguous
__global__ void k_w_to_f16t(const float* W, _Float16* Wt) {
    int i = blockIdx.x * blockDim.x + threadIdx.x;
    if (i < FD * FD) {
        int r = i >> 8, c = i & 255;          // r = K row, c = N col
        Wt[c * FD + r] = (_Float16)W[i];
    }
}

// ---------- WMMA GEMM: D[M,256] = A16[M,256] * (Bt16[n,k])^T ----------
// grid.x = M/16 blocks, 256 threads (8 waves). Wave w covers cols [32w, 32w+32).
__global__ void k_gemm_wmma(const _Float16* __restrict__ A,
                            const _Float16* __restrict__ Bt,
                            float* __restrict__ D, int M) {
    const int lane = threadIdx.x & 31;
    const int wave = threadIdx.x >> 5;
    const int mbase = blockIdx.x * 16;
    const int nbase = wave * 32;

    const int row  = mbase + (lane & 15);
    const int krow = (lane >> 4) * 8;          // 0 or 8
    const int n0   = nbase + (lane & 15);
    const int n1   = n0 + 16;

    v8f c0 = {}; v8f c1 = {};
    const _Float16* arow  = A  + (long)row * FD;
    const _Float16* brow0 = Bt + (long)n0  * FD;
    const _Float16* brow1 = Bt + (long)n1  * FD;

    #pragma unroll
    for (int k0 = 0; k0 < FD; k0 += 32) {
        int kb = k0 + krow;
        v16h a  = load_frag16(arow  + kb);
        v16h b0 = load_frag16(brow0 + kb);
        v16h b1 = load_frag16(brow1 + kb);
        __builtin_prefetch(arow + kb + 32, 0, 1);
        c0 = __builtin_amdgcn_wmma_f32_16x16x32_f16(false, a, false, b0, (short)0, c0, false, false);
        c1 = __builtin_amdgcn_wmma_f32_16x16x32_f16(false, a, false, b1, (short)0, c1, false, false);
    }

    // C/D layout: lane&15 = N col, VGPR i = M row (lane>>4)*8 + i
    const int colA  = nbase + (lane & 15);
    const int colB  = colA + 16;
    const int rbase = mbase + (lane >> 4) * 8;
    #pragma unroll
    for (int i = 0; i < 8; ++i) {
        D[(long)(rbase + i) * FD + colA] = c0[i];
        D[(long)(rbase + i) * FD + colB] = c1[i];
    }
}

// ---------- attention coefficients: a_src/a_dst [N,H] ----------
__global__ void k_attn_coef(const float* __restrict__ h,
                            const float* __restrict__ att_s,
                            const float* __restrict__ att_d,
                            float* __restrict__ asrc, float* __restrict__ adst) {
    int idx = blockIdx.x * blockDim.x + threadIdx.x;
    if (idx >= NN * HH) return;
    int n = idx >> 2, hd = idx & 3;
    const float* hp = h + (long)n * FD + hd * CC;
    const float* ws = att_s + hd * CC;
    const float* wd = att_d + hd * CC;
    float ss = 0.f, sd = 0.f;
    #pragma unroll 8
    for (int c = 0; c < CC; ++c) { float v = hp[c]; ss += v * ws[c]; sd += v * wd[c]; }
    asrc[idx] = ss; adst[idx] = sd;
}

__device__ __forceinline__ void edge_sd(const long long* ei, int e, int& s, int& d) {
    if (e < EEDGE) { s = (int)ei[e]; d = (int)ei[EEDGE + e]; }
    else           { s = e - EEDGE; d = e - EEDGE; }
}

// ---------- edge pass A: segment max ----------
__global__ void k_edge_max(const long long* __restrict__ ei,
                           const float* __restrict__ asrc, const float* __restrict__ adst,
                           float* __restrict__ mmax) {
    int idx = blockIdx.x * blockDim.x + threadIdx.x;
    if (idx >= ETOT * HH) return;
    int e = idx >> 2, hd = idx & 3, s, d;
    edge_sd(ei, e, s, d);
    float v = asrc[s * HH + hd] + adst[d * HH + hd];
    v = v > 0.f ? v : NEGSL * v;
    atomicMaxF(&mmax[d * HH + hd], v);
}

// ---------- edge pass B: exp + segment sum ----------
__global__ void k_edge_exp(const long long* __restrict__ ei,
                           const float* __restrict__ asrc, const float* __restrict__ adst,
                           const float* __restrict__ mmax,
                           float* __restrict__ exbuf, float* __restrict__ ssum) {
    int idx = blockIdx.x * blockDim.x + threadIdx.x;
    if (idx >= ETOT * HH) return;
    int e = idx >> 2, hd = idx & 3, s, d;
    edge_sd(ei, e, s, d);
    float v = asrc[s * HH + hd] + adst[d * HH + hd];
    v = v > 0.f ? v : NEGSL * v;
    float ex = __expf(v - mmax[d * HH + hd]);
    exbuf[idx] = ex;
    atomicAdd(&ssum[d * HH + hd], ex);
}

// ---------- edge pass C: out[dst] += alpha * h[src] ----------
__global__ void k_edge_agg(const long long* __restrict__ ei,
                           const float* __restrict__ exbuf, const float* __restrict__ ssum,
                           const float* __restrict__ h, float* __restrict__ out) {
    unsigned idx = blockIdx.x * blockDim.x + threadIdx.x;   // ETOT*256 items
    int e = idx >> 8, c = idx & 255, hd = c >> 6, s, d;
    edge_sd(ei, e, s, d);
    float alpha = exbuf[e * HH + hd] / ssum[d * HH + hd];
    atomicAdd(&out[(long)d * FD + c], alpha * h[(long)s * FD + c]);
}

// ---------- post: out = elu(agg + b) + residual ----------
__global__ void k_post(float* __restrict__ agg, const float* __restrict__ b,
                       const float* __restrict__ res) {
    long idx = (long)blockIdx.x * blockDim.x + threadIdx.x;
    if (idx >= (long)NN * FD) return;
    int c = (int)(idx & 255);
    float v = agg[idx] + b[c];
    v = v > 0.f ? v : (__expf(v) - 1.0f);
    agg[idx] = v + res[idx];
}

// ---------- batchnorm: per-column stats then affine ----------
__global__ void k_bnstats(const float* __restrict__ X, float* __restrict__ sum,
                          float* __restrict__ sumsq) {
    int r0 = blockIdx.x * 128;
    int c  = threadIdx.x;
    float s = 0.f, q = 0.f;
    int rend = r0 + 128; if (rend > NN) rend = NN;
    for (int r = r0; r < rend; ++r) {
        float v = X[(long)r * FD + c];
        s += v; q += v * v;
    }
    atomicAdd(&sum[c], s);
    atomicAdd(&sumsq[c], q);
}

__global__ void k_bnfinal(const float* __restrict__ sum, const float* __restrict__ sumsq,
                          const float* __restrict__ gamma, const float* __restrict__ beta,
                          float* __restrict__ scale, float* __restrict__ shift) {
    int c = threadIdx.x;     // 256 threads
    float mu  = sum[c]  * (1.0f / NN);
    float var = sumsq[c] * (1.0f / NN) - mu * mu;
    float inv = rsqrtf(var + BNEPS);
    float sc  = gamma[c] * inv;
    scale[c] = sc;
    shift[c] = beta[c] - sc * mu;
}

__global__ void k_bnapply(const float* __restrict__ X, const float* __restrict__ scale,
                          const float* __restrict__ shift, float* __restrict__ Y) {
    long idx = (long)blockIdx.x * blockDim.x + threadIdx.x;
    if (idx >= (long)NN * FD) return;
    int c = (int)(idx & 255);
    Y[idx] = X[idx] * scale[c] + shift[c];
}

// ---------- pooling ----------
__global__ void k_cnt(const long long* __restrict__ batch, float* __restrict__ cnt) {
    int n = blockIdx.x * blockDim.x + threadIdx.x;
    if (n < NN) atomicAdd(&cnt[(int)batch[n]], 1.0f);
}
__global__ void k_pool(const long long* __restrict__ batch, const float* __restrict__ h,
                       float* __restrict__ pooled) {
    long idx = (long)blockIdx.x * blockDim.x + threadIdx.x;
    if (idx >= (long)NN * FD) return;
    int n = (int)(idx >> 8), c = (int)(idx & 255);
    atomicAdd(&pooled[(int)batch[n] * FD + c], h[idx]);
}
__global__ void k_pooldiv(float* __restrict__ pooled, const float* __restrict__ cnt) {
    int idx = blockIdx.x * blockDim.x + threadIdx.x;
    if (idx >= GG * FD) return;
    float c = cnt[idx >> 8];
    pooled[idx] /= (c > 1.0f ? c : 1.0f);
}

// ---------- MLP head ----------
__global__ void k_mlp1(const float* __restrict__ pooled, const float* __restrict__ w,
                       const float* __restrict__ b, float* __restrict__ z) {
    int idx = blockIdx.x * blockDim.x + threadIdx.x;   // GG*64
    if (idx >= GG * CC) return;
    int g = idx >> 6, j = idx & 63;
    const float* pr = pooled + g * FD;
    float acc = b[j];
    #pragma unroll 8
    for (int k = 0; k < FD; ++k) acc += pr[k] * w[k * CC + j];
    z[idx] = acc > 0.f ? acc : 0.f;
}

__global__ void k_mlp2(const float* __restrict__ z, const float* __restrict__ w,
                       const float* __restrict__ b, float* __restrict__ out) {
    __shared__ float sl[NCLS];
    __shared__ float lse;
    int g = blockIdx.x, j = threadIdx.x;
    if (j < NCLS) {
        const float* zr = z + g * CC;
        float acc = b[j];
        #pragma unroll
        for (int k = 0; k < CC; ++k) acc += zr[k] * w[k * NCLS + j];
        sl[j] = acc;
    }
    __syncthreads();
    if (j == 0) {
        float m = sl[0];
        for (int k = 1; k < NCLS; ++k) m = sl[k] > m ? sl[k] : m;
        float se = 0.f;
        for (int k = 0; k < NCLS; ++k) se += __expf(sl[k] - m);
        lse = m + __logf(se);
    }
    __syncthreads();
    if (j < NCLS) out[g * NCLS + j] = sl[j] - lse;
}

// ---------- orchestration ----------
extern "C" void kernel_launch(void* const* d_in, const int* in_sizes, int n_in,
                              void* d_out, int out_size, void* d_ws, size_t ws_size,
                              hipStream_t stream) {
    const float*     x    = (const float*)d_in[0];
    const long long* ei   = (const long long*)d_in[1];
    const long long* bat  = (const long long*)d_in[2];
    const float *W1 = (const float*)d_in[3],  *as1 = (const float*)d_in[4],
                *ad1 = (const float*)d_in[5], *b1 = (const float*)d_in[6],
                *g1 = (const float*)d_in[7],  *be1 = (const float*)d_in[8];
    const float *W2 = (const float*)d_in[9],  *as2 = (const float*)d_in[10],
                *ad2 = (const float*)d_in[11],*b2 = (const float*)d_in[12],
                *g2 = (const float*)d_in[13], *be2 = (const float*)d_in[14];
    const float *f1w = (const float*)d_in[15], *f1b = (const float*)d_in[16];
    const float *f2w = (const float*)d_in[17], *f2b = (const float*)d_in[18];
    float* out = (float*)d_out;

    char* w = (char*)d_ws; size_t off = 0;
    auto carve = [&](size_t bytes) -> void* {
        void* p = w + off; off = (off + bytes + 511) & ~(size_t)511; return p;
    };
    _Float16* W1t  = (_Float16*)carve((size_t)FD * FD * 2);
    _Float16* W2t  = (_Float16*)carve((size_t)FD * FD * 2);
    _Float16* xin  = (_Float16*)carve((size_t)NN * FD * 2);
    float* hlin  = (float*)carve((size_t)NN * FD * 4);
    float* hcur  = (float*)carve((size_t)NN * FD * 4);
    float* hout  = (float*)carve((size_t)NN * FD * 4);
    float* asrc  = (float*)carve((size_t)NN * HH * 4);
    float* adst  = (float*)carve((size_t)NN * HH * 4);
    float* mmax  = (float*)carve((size_t)NN * HH * 4);
    float* ssum  = (float*)carve((size_t)NN * HH * 4);
    float* exbuf = (float*)carve((size_t)ETOT * HH * 4);
    float* csum  = (float*)carve(FD * 4);
    float* csq   = (float*)carve(FD * 4);
    float* scale = (float*)carve(FD * 4);
    float* shift = (float*)carve(FD * 4);
    float* pooled= (float*)carve((size_t)GG * FD * 4);
    float* cnt   = (float*)carve(GG * 4);
    float* zbuf  = (float*)carve((size_t)GG * CC * 4);

    const int T = 256;
    const long NF = (long)NN * FD;
    const int gNF   = (int)((NF + T - 1) / T);
    const int gNH   = (NN * HH + T - 1) / T;
    const int gEH   = (ETOT * HH + T - 1) / T;
    const int gEF   = ETOT;                     // ETOT*256 / 256
    const int gGemm = NN / 16;                  // 3125 M-tiles
    const int gBn   = (NN + 127) / 128;

    // weight prep
    k_w_to_f16t<<<(FD * FD + T - 1) / T, T, 0, stream>>>(W1, W1t);
    k_w_to_f16t<<<(FD * FD + T - 1) / T, T, 0, stream>>>(W2, W2t);

    auto gat_layer = [&](const float* in_f32, const _Float16* Wt,
                         const float* att_s, const float* att_d, const float* bias,
                         const float* res) {
        k_f32_to_f16<<<gNF, T, 0, stream>>>(in_f32, xin, NF);
        k_gemm_wmma<<<gGemm, T, 0, stream>>>(xin, Wt, hlin, NN);
        k_attn_coef<<<gNH, T, 0, stream>>>(hlin, att_s, att_d, asrc, adst);
        k_fill<<<gNF, T, 0, stream>>>(hout, NF, 0.f);
        k_fill<<<gNH, T, 0, stream>>>(ssum, (long)NN * HH, 0.f);
        k_fill<<<gNH, T, 0, stream>>>(mmax, (long)NN * HH, -3.0e38f);
        k_edge_max<<<gEH, T, 0, stream>>>(ei, asrc, adst, mmax);
        k_edge_exp<<<gEH, T, 0, stream>>>(ei, asrc, adst, mmax, exbuf, ssum);
        k_edge_agg<<<gEF, T, 0, stream>>>(ei, exbuf, ssum, hlin, hout);
        k_post<<<gNF, T, 0, stream>>>(hout, bias, res);
    };

    auto bn = [&](const float* gmm, const float* bet, float* dst) {
        k_fill<<<1, T, 0, stream>>>(csum, FD, 0.f);
        k_fill<<<1, T, 0, stream>>>(csq,  FD, 0.f);
        k_bnstats<<<gBn, T, 0, stream>>>(hout, csum, csq);
        k_bnfinal<<<1, FD, 0, stream>>>(csum, csq, gmm, bet, scale, shift);
        k_bnapply<<<gNF, T, 0, stream>>>(hout, scale, shift, dst);
    };

    // layer 1: residual is original x; BN1 output -> hcur
    gat_layer(x, W1t, as1, ad1, b1, x);
    bn(g1, be1, hcur);

    // layer 2: input & residual = hcur; BN2 output -> hout (in place)
    gat_layer(hcur, W2t, as2, ad2, b2, hcur);
    bn(g2, be2, hout);

    // global mean pool + MLP head
    k_fill<<<(GG * FD + T - 1) / T, T, 0, stream>>>(pooled, (long)GG * FD, 0.f);
    k_fill<<<1, GG, 0, stream>>>(cnt, GG, 0.f);
    k_cnt<<<(NN + T - 1) / T, T, 0, stream>>>(bat, cnt);
    k_pool<<<gNF, T, 0, stream>>>(bat, hout, pooled);
    k_pooldiv<<<(GG * FD + T - 1) / T, T, 0, stream>>>(pooled, cnt);
    k_mlp1<<<(GG * CC + T - 1) / T, T, 0, stream>>>(pooled, f1w, f1b, zbuf);
    k_mlp2<<<GG, 32, 0, stream>>>(zbuf, f2w, f2b, out);
}
